// ProtoCount_3633542332975
// MI455X (gfx1250) — compile-verified, matrix operand order checked by reading, not run on hardware
//
#include <hip/hip_runtime.h>
#include <stdint.h>

typedef __attribute__((ext_vector_type(16))) __bf16 v16bf;
typedef __attribute__((ext_vector_type(2)))  __bf16 v2bf;
typedef __attribute__((ext_vector_type(8)))  float  v8f;
typedef __attribute__((ext_vector_type(4)))  float  f4v;

#define D_DIM 1024
#define P_DIM 512
#define NTILE 64      // columns of x per workgroup
#define KCH   32      // K chunk per iteration (one WMMA K)
#define NIT   (D_DIM / KCH)
#define XPAD  40      // padded LDS row stride (uint16 units) -> conflict-free

// fp32 pair -> packed bf16 (RNE) in one VALU op (v_cvt_pk_bf16_f32).
__device__ __forceinline__ uint32_t pkbf(float lo, float hi) {
#if defined(__has_builtin) && __has_builtin(__builtin_amdgcn_cvt_pk_bf16_f32)
    return __builtin_bit_cast(uint32_t, __builtin_amdgcn_cvt_pk_bf16_f32(lo, hi));
#else
    v2bf h;
    h.x = (__bf16)lo;
    h.y = (__bf16)hi;
    return __builtin_bit_cast(uint32_t, h);
#endif
}

__device__ __forceinline__ v16bf ld16bf(const uint16_t* p) {
    v16bf v;
    __builtin_memcpy(&v, p, sizeof(v));
    return v;
}

// gfx1250 async global->LDS copy, tracked by ASYNCcnt (no VGPR data path).
__device__ __forceinline__ void async_copy_b128(uint32_t lds_off, const void* gptr) {
    asm volatile("global_load_async_to_lds_b128 %0, %1, off"
                 :: "v"(lds_off), "v"((uint64_t)(uintptr_t)gptr)
                 : "memory");
}
__device__ __forceinline__ void wait_async_zero() {
    asm volatile("s_wait_asynccnt 0x0" ::: "memory");
}

// ---------------------------------------------------------------------------
// Kernel 1: prototypes fp32 -> bf16, p_sq (fp32), zero histogram bins.
// grid = P, block = 256. Each block handles one prototype row (1024 floats).
// ---------------------------------------------------------------------------
__global__ __launch_bounds__(256) void ProtoPrep(const float* __restrict__ protos,
                                                 uint16_t* __restrict__ pbf,
                                                 float* __restrict__ psq,
                                                 unsigned* __restrict__ counts) {
    __shared__ float wsum[8];
    const int row = blockIdx.x;
    const int t   = threadIdx.x;

    const f4v v = *(const f4v*)(protos + (size_t)row * D_DIM + t * 4);
    uint2 pk;
    pk.x = pkbf(v.x, v.y);
    pk.y = pkbf(v.z, v.w);
    *(uint2*)(pbf + (size_t)row * D_DIM + t * 4) = pk;

    float ss = v.x * v.x + v.y * v.y + v.z * v.z + v.w * v.w;
    #pragma unroll
    for (int o = 16; o > 0; o >>= 1) ss += __shfl_xor(ss, o);
    if ((t & 31) == 0) wsum[t >> 5] = ss;
    __syncthreads();
    if (t == 0) {
        float s = 0.f;
        #pragma unroll
        for (int i = 0; i < 8; ++i) s += wsum[i];
        psq[row]    = s;
        counts[row] = 0u;   // zero histogram (grid == P)
    }
}

// ---------------------------------------------------------------------------
// Kernel 2: fused GEMM (bf16 WMMA) + argmin over P + histogram atomics.
// grid = N/64, block = 256 (8 waves). Wave w owns P rows [w*64, w*64+64).
// Double-buffered LDS; prototype chunks staged with async global->LDS b128.
// ---------------------------------------------------------------------------
__global__ __launch_bounds__(256) void ProtoAssign(const float* __restrict__ x,
                                                   const uint16_t* __restrict__ pbf,
                                                   const float* __restrict__ psq,
                                                   unsigned* __restrict__ counts) {
    __shared__ uint16_t lds_p[2][P_DIM][XPAD];   // 2 x (512 x 32 bf16) chunks
    __shared__ uint16_t lds_x[2][NTILE][XPAD];   // 2 x (64 x 32 bf16) chunks
    __shared__ float    lds_psq[P_DIM];
    __shared__ unsigned long long lds_pack[NTILE];

    const int t    = threadIdx.x;
    const int lane = t & 31;
    const int wave = t >> 5;
    const int col  = lane & 15;
    const int half = lane >> 4;
    const size_t nBase = (size_t)blockIdx.x * NTILE;

    lds_psq[t]       = psq[t];
    lds_psq[t + 256] = psq[t + 256];
    if (t < NTILE) lds_pack[t] = ~0ull;

    const int xrow = t >> 2;   // 0..63
    const int xq   = t & 3;    // quarter of the 32-wide K chunk

    // stage prototype chunk k0 -> lds_p[buf] via async copies (8 x b128/thread)
    auto stageP = [&](int buf, int k0) {
        #pragma unroll
        for (int i = 0; i < 8; ++i) {
            const int lin = i * 256 + t;
            const int row = lin >> 2;
            const int q   = lin & 3;
            async_copy_b128((uint32_t)(uintptr_t)&lds_p[buf][row][q * 8],
                            pbf + (size_t)row * D_DIM + k0 + q * 8);
        }
    };
    // stage x chunk k0 -> lds_x[buf]: NT fp32 load, packed bf16 cvt, b128 store
    auto stageX = [&](int buf, int k0) {
        const float* xs = x + (nBase + (size_t)xrow) * D_DIM + k0 + xq * 8;
        const f4v a = __builtin_nontemporal_load((const f4v*)xs);
        const f4v b = __builtin_nontemporal_load((const f4v*)xs + 1);
        uint4 xp;
        xp.x = pkbf(a.x, a.y);
        xp.y = pkbf(a.z, a.w);
        xp.z = pkbf(b.x, b.y);
        xp.w = pkbf(b.z, b.w);
        *(uint4*)&lds_x[buf][xrow][xq * 8] = xp;
    };

    const v8f vzero = {0.f, 0.f, 0.f, 0.f, 0.f, 0.f, 0.f, 0.f};
    v8f acc[4][4];   // [p-subtile][n-subtile], 16x16 f32 each
    #pragma unroll
    for (int pi = 0; pi < 4; ++pi)
        #pragma unroll
        for (int ni = 0; ni < 4; ++ni) acc[pi][ni] = vzero;

    // prologue: fill buffer 0
    stageP(0, 0);
    stageX(0, 0);
    wait_async_zero();
    __syncthreads();

    for (int it = 0; it < NIT; ++it) {
        const int buf = it & 1;

        // (a) issue this chunk's fragment loads from LDS
        v16bf Bf[4], Af[4];
        #pragma unroll
        for (int ni = 0; ni < 4; ++ni)
            Bf[ni] = ld16bf(&lds_x[buf][ni * 16 + col][half * 16]);
        #pragma unroll
        for (int pi = 0; pi < 4; ++pi)
            Af[pi] = ld16bf(&lds_p[buf][wave * 64 + pi * 16 + col][half * 16]);

        // (b) stage next chunk into the other buffer (overlaps with WMMAs)
        if (it + 1 < NIT) {
            stageP(buf ^ 1, (it + 1) * KCH);
            stageX(buf ^ 1, (it + 1) * KCH);
        }

        // (c) 16 WMMAs: 64P x 64N x 32K per wave
        #pragma unroll
        for (int pi = 0; pi < 4; ++pi)
            #pragma unroll
            for (int ni = 0; ni < 4; ++ni)
                acc[pi][ni] = __builtin_amdgcn_wmma_f32_16x16x32_bf16(
                    false, Af[pi], false, Bf[ni], (short)0, acc[pi][ni], false, false);

        // (d) close the pipeline stage
        wait_async_zero();
        __syncthreads();
    }

    // --- epilogue: score = p_sq - 2*dot ; argmin over this wave's 64 rows ---
    #pragma unroll
    for (int ni = 0; ni < 4; ++ni) {
        float mv = 3.402823466e+38f;
        int   mi = 0;
        #pragma unroll
        for (int pi = 0; pi < 4; ++pi) {
            const int pb = wave * 64 + pi * 16 + half * 8;
            #pragma unroll
            for (int r = 0; r < 8; ++r) {
                const float s = lds_psq[pb + r] - 2.0f * acc[pi][ni][r];
                if (s < mv) { mv = s; mi = pb + r; }
            }
        }
        // combine the two lane-halves (same column n, different M rows)
        const float ov = __shfl_xor(mv, 16);
        const int   oi = __shfl_xor(mi, 16);
        if (ov < mv || (ov == mv && oi < mi)) { mv = ov; mi = oi; }

        // sortable-uint key || index ; ds_min_u64 merges across waves,
        // ties resolve to smallest index (matches argmin semantics)
        unsigned key = __float_as_uint(mv);
        key = (key & 0x80000000u) ? ~key : (key | 0x80000000u);
        const unsigned long long pk =
            ((unsigned long long)key << 32) | (unsigned)mi;
        atomicMin(&lds_pack[ni * 16 + col], pk);
    }
    __syncthreads();

    if (t < NTILE) {
        const unsigned idx = (unsigned)(lds_pack[t] & 0xFFFFFFFFull);
        atomicAdd(&counts[idx], 1u);
    }
}

// ---------------------------------------------------------------------------
// Kernel 3: L2-normalize histogram. 1 block x 512 threads.
// ---------------------------------------------------------------------------
__global__ __launch_bounds__(512) void ProtoNorm(const unsigned* __restrict__ counts,
                                                 float* __restrict__ out) {
    __shared__ float s[P_DIM];
    const int t = threadIdx.x;
    const float c = (float)counts[t];
    s[t] = c * c;
    __syncthreads();
    #pragma unroll
    for (int o = 256; o > 0; o >>= 1) {
        if (t < o) s[t] += s[t + o];
        __syncthreads();
    }
    out[t] = c / sqrtf(s[0]);
}

// ---------------------------------------------------------------------------
extern "C" void kernel_launch(void* const* d_in, const int* in_sizes, int n_in,
                              void* d_out, int out_size, void* d_ws, size_t ws_size,
                              hipStream_t stream) {
    const float* x      = (const float*)d_in[0];
    const float* protos = (const float*)d_in[1];
    const int N = in_sizes[0] / D_DIM;   // 131072
    const int P = in_sizes[1] / D_DIM;   // 512

    uint16_t* pbf    = (uint16_t*)d_ws;                                   // P*D bf16
    float*    psq    = (float*)((char*)d_ws + (size_t)P * D_DIM * 2);     // P floats
    unsigned* counts = (unsigned*)((char*)psq + (size_t)P * sizeof(float));
    float*    out    = (float*)d_out;

    ProtoPrep<<<P, 256, 0, stream>>>(protos, pbf, psq, counts);
    ProtoAssign<<<N / NTILE, 256, 0, stream>>>(x, pbf, psq, counts);
    ProtoNorm<<<1, P, 0, stream>>>(counts, out);
}
